// Q_JetTagger_472446403061
// MI455X (gfx1250) — compile-verified, matrix operand order checked by reading, not run on hardware
//
#include <hip/hip_runtime.h>
#include <hip/hip_bf16.h>

// ---------------------------------------------------------------------------
// HAWQ quantized 4-layer MLP (16->64->32->32->5) + softmax on gfx1250.
// Integer GEMM done EXACTLY via V_WMMA_I32_16X16X64_IU8 with the 16-bit
// activation split into q = 256*q_hi(i8) + q_lo(u8)  (two WMMAs per tile).
// Per-tensor act scales require global reductions between layers, so we use a
// recompute strategy: stage k recomputes layers 1..k and emits only the max.
// x (64MB) stays resident in the 192MB L2, so re-reads are ~free; HBM traffic
// is ~64MB in + 20MB out.
// ---------------------------------------------------------------------------

typedef int v8i __attribute__((ext_vector_type(8)));

#define N_ACT 32767.0f          // 2^(16-1)-1
#define ROWS  1048576
#define TILES (ROWS / 16)       // 65536 row tiles of 16

struct WS {
    unsigned   aw[4];           // float-bits of absmax: |x|, relu(h1), relu(h2), relu(h3)
    float      wscale[4][64];   // per-channel weight scales (padded)
    float      bias[4][64];     // raw biases (padded with 0)
    signed char wq[4][64][64];  // int8 weights [layer][outCh pad 64][K pad 64], zero padded
};

// K-group base for VGPR v of an 8-bit 16x64 A fragment (mirrored for B),
// per CDNA5 ISA 7.12.2: V0: K0-3 / K8-11, V1: K4-7 / K12-15, V2-3: +16, V4-7: +32.
__device__ __forceinline__ int kbase(int v, int g) {
    return ((v >> 1) << 4) + (g << 3) + ((v & 1) << 2);
}

__device__ __forceinline__ int pack_lo(int a, int b, int c, int d) {
    return (a & 255) | ((b & 255) << 8) | ((c & 255) << 16) | ((d & 255) << 24);
}
__device__ __forceinline__ int pack_hi(int a, int b, int c, int d) {
    return ((a >> 8) & 255) | (((b >> 8) & 255) << 8) |
           (((c >> 8) & 255) << 16) | (((d >> 8) & 255) << 24);
}

__device__ __forceinline__ int qclamp16(float x, float inv) {
    float r = rintf(x * inv);                     // jnp.round == RNE
    r = fminf(fmaxf(r, -N_ACT), N_ACT);
    return (int)r;
}

__device__ __forceinline__ void block_max_atomic(float mx, float* sred, unsigned* dst) {
    const int t = threadIdx.x;
    sred[t] = mx;
    __syncthreads();
    #pragma unroll
    for (int s = 128; s > 0; s >>= 1) {
        if (t < s) sred[t] = fmaxf(sred[t], sred[t + s]);
        __syncthreads();
    }
    if (t == 0) atomicMax(dst, __float_as_uint(sred[0]));  // vals >= 0 -> bits monotonic
}

// Load one 64x16 int8 B fragment (N tile t of layer L) + dequant constants.
__device__ __forceinline__ void load_B(const WS* __restrict__ ws, int L, int t,
                                       int g, int nIdx, float s_in,
                                       v8i& B, float& cs, float& bq) {
    const int ch = t * 16 + nIdx;
    const signed char* wrow = ws->wq[L][ch];
    #pragma unroll
    for (int v = 0; v < 8; ++v)
        B[v] = *(const int*)(wrow + kbase(v, g));
    const float wsc = ws->wscale[L][ch];
    cs = wsc * s_in;                               // == bias_scale == dequant scale
    const float b = ws->bias[L][ch];
    bq = (cs > 0.0f) ? rintf(b / cs) * cs : 0.0f;  // 32-bit bias fake-quant
}

// Rebuild hi/lo A fragments from the int16 activation row buffer in LDS.
__device__ __forceinline__ void load_A(const short* act, int M, int g, v8i& lo, v8i& hi) {
    #pragma unroll
    for (int v = 0; v < 8; ++v) {
        const short4 s = *(const short4*)(act + M * 64 + kbase(v, g));
        const int q0 = s.x, q1 = s.y, q2 = s.z, q3 = s.w;
        lo[v] = pack_lo(q0, q1, q2, q3);
        hi[v] = pack_hi(q0, q1, q2, q3);
    }
}

// One layer: NT output N-tiles, exact int32 GEMM via split-iu8 WMMA, dequant+bias(+relu).
template <int NT, bool RELU>
__device__ __forceinline__ void run_layer(const v8i& Alo, const v8i& Ahi,
                                          const v8i (&B)[NT], const float (&cs)[NT],
                                          const float (&bq)[NT], float (&h)[NT][8]) {
    #pragma unroll
    for (int t = 0; t < NT; ++t) {
        v8i z = {};
        v8i dlo = __builtin_amdgcn_wmma_i32_16x16x64_iu8(false, Alo, true, B[t], z, false, false);
        v8i dhi = __builtin_amdgcn_wmma_i32_16x16x64_iu8(true,  Ahi, true, B[t], z, false, false);
        v8i dt  = (dhi << 8) + dlo;                // exact: q = 256*q_hi + q_lo
        #pragma unroll
        for (int i = 0; i < 8; ++i) {
            float v = (float)dt[i] * cs[t] + bq[t];
            h[t][i] = RELU ? fmaxf(v, 0.0f) : v;
        }
    }
}

// Quantize layer output (D-frag layout) to int16 and scatter to LDS act[16][64].
template <int NT>
__device__ __forceinline__ void quant_store(const float (&h)[NT][8], float inv,
                                            short* act, int g, int nIdx) {
    #pragma unroll
    for (int t = 0; t < NT; ++t)
        #pragma unroll
        for (int i = 0; i < 8; ++i) {
            const int row = i + (g << 3);          // D frag: VGPR i -> M = i (+8 for hi half)
            const int col = t * 16 + nIdx;
            act[row * 64 + col] = (short)qclamp16(h[t][i], inv);
        }
}

template <int NT>
__device__ __forceinline__ void accmax(const float (&h)[NT][8], float& mx) {
    #pragma unroll
    for (int t = 0; t < NT; ++t)
        #pragma unroll
        for (int i = 0; i < 8; ++i) mx = fmaxf(mx, h[t][i]);
}

// ---------------------------------------------------------------------------
__global__ void init_kernel(WS* ws) {
    if (threadIdx.x < 4) ws->aw[threadIdx.x] = 0u;
}

__global__ __launch_bounds__(256) void prep_kernel(
        const float* __restrict__ W1, const float* __restrict__ b1,
        const float* __restrict__ W2, const float* __restrict__ b2,
        const float* __restrict__ W3, const float* __restrict__ b3,
        const float* __restrict__ W4, const float* __restrict__ b4,
        WS* __restrict__ ws) {
    const float* Wp[4] = {W1, W2, W3, W4};
    const float* bp[4] = {b1, b2, b3, b4};
    const int Kr[4] = {16, 64, 32, 32};
    const int Or[4] = {64, 32, 32, 5};
    const int idx = threadIdx.x;                   // 256 threads = 4 layers x 64 channels
    const int L = idx >> 6, ch = idx & 63;
    signed char* wrow = ws->wq[L][ch];
    if (ch < Or[L]) {
        const float* wr = Wp[L] + ch * Kr[L];
        float m = 0.0f;
        for (int k = 0; k < Kr[L]; ++k) m = fmaxf(m, fabsf(wr[k]));
        const float sc  = m / 127.0f;
        const float inv = (m > 0.0f) ? 127.0f / m : 0.0f;
        for (int k = 0; k < 64; ++k) {
            float q = (k < Kr[L]) ? fminf(fmaxf(rintf(wr[k] * inv), -127.0f), 127.0f) : 0.0f;
            wrow[k] = (signed char)(int)q;
        }
        ws->wscale[L][ch] = sc;
        ws->bias[L][ch]   = bp[L][ch];
    } else {
        for (int k = 0; k < 64; ++k) wrow[k] = 0;
        ws->wscale[L][ch] = 0.0f;
        ws->bias[L][ch]   = 0.0f;
    }
}

__global__ __launch_bounds__(256) void absmax_kernel(const float4* __restrict__ x,
                                                     int n4, WS* __restrict__ ws) {
    __shared__ float sred[256];
    float mx = 0.0f;
    for (int i = blockIdx.x * blockDim.x + threadIdx.x; i < n4;
         i += gridDim.x * blockDim.x) {
        const float4 f = x[i];
        mx = fmaxf(mx, fmaxf(fmaxf(fabsf(f.x), fabsf(f.y)),
                             fmaxf(fabsf(f.z), fabsf(f.w))));
    }
    block_max_atomic(mx, sred, &ws->aw[0]);
}

// Stage kernel: computes layers 1..NL. NL<4: atomic-max relu output into aw[NL].
// NL==4: softmax, write output. One wave per 16-row tile, grid-stride.
template <int NL>
__global__ __launch_bounds__(256) void fwd_kernel(const float* __restrict__ x,
                                                  float* __restrict__ out,
                                                  WS* __restrict__ ws) {
    __shared__ __align__(16) unsigned char sbuf[8 * 3072];  // per wave: 2KB act + 1KB softmax
    __shared__ float sred[256];

    const int lane = threadIdx.x & 31;
    const int wid  = threadIdx.x >> 5;
    const int g    = lane >> 4;       // half-wave selector
    const int nIdx = lane & 15;       // N / M index within frag
    short* act  = (short*)(sbuf + wid * 3072);
    float* hbuf = (float*)(sbuf + wid * 3072 + 2048);

    float sact[NL], invs[NL];
    #pragma unroll
    for (int j = 0; j < NL; ++j) {
        const float a = __uint_as_float(ws->aw[j]);
        sact[j] = a * (1.0f / N_ACT);
        invs[j] = (a > 0.0f) ? (N_ACT / a) : 0.0f;
    }

    // Resident B fragments + dequant constants (9 frags max = 72 VGPRs, L2-cached loads).
    v8i B1[4]; float cs1[4], bq1[4];
    v8i B2[2]; float cs2[2], bq2[2];
    v8i B3[2]; float cs3[2], bq3[2];
    v8i B4[1]; float cs4[1], bq4[1];
    #pragma unroll
    for (int t = 0; t < 4; ++t) load_B(ws, 0, t, g, nIdx, sact[0], B1[t], cs1[t], bq1[t]);
    if constexpr (NL >= 2) {
        #pragma unroll
        for (int t = 0; t < 2; ++t) load_B(ws, 1, t, g, nIdx, sact[1], B2[t], cs2[t], bq2[t]);
    }
    if constexpr (NL >= 3) {
        #pragma unroll
        for (int t = 0; t < 2; ++t) load_B(ws, 2, t, g, nIdx, sact[2], B3[t], cs3[t], bq3[t]);
    }
    if constexpr (NL >= 4) load_B(ws, 3, 0, g, nIdx, sact[3], B4[0], cs4[0], bq4[0]);

    const int nWaves = gridDim.x << 3;
    const int waveId = (blockIdx.x << 3) + wid;
    float mx = 0.0f;

    for (int tile = waveId; tile < TILES; tile += nWaves) {
        const int row0 = tile << 4;

        // Layer-1 A frag straight from global: lane holds row nIdx, K = g*8..g*8+7 (K 16..63 zero).
        v8i Alo = {}; v8i Ahi = {};
        {
            const float4* xr = (const float4*)(x + (size_t)(row0 + nIdx) * 16 + (g << 3));
            const float4 f0 = xr[0];
            const float4 f1 = xr[1];
            const int q0 = qclamp16(f0.x, invs[0]), q1 = qclamp16(f0.y, invs[0]);
            const int q2 = qclamp16(f0.z, invs[0]), q3 = qclamp16(f0.w, invs[0]);
            const int q4 = qclamp16(f1.x, invs[0]), q5 = qclamp16(f1.y, invs[0]);
            const int q6 = qclamp16(f1.z, invs[0]), q7 = qclamp16(f1.w, invs[0]);
            Alo[0] = pack_lo(q0, q1, q2, q3); Ahi[0] = pack_hi(q0, q1, q2, q3);
            Alo[1] = pack_lo(q4, q5, q6, q7); Ahi[1] = pack_hi(q4, q5, q6, q7);
        }

        float h1[4][8];
        run_layer<4, true>(Alo, Ahi, B1, cs1, bq1, h1);
        if constexpr (NL == 1) { accmax<4>(h1, mx); }
        else {
            quant_store<4>(h1, invs[1], act, g, nIdx);
            load_A(act, nIdx, g, Alo, Ahi);
            float h2[2][8];
            run_layer<2, true>(Alo, Ahi, B2, cs2, bq2, h2);
            if constexpr (NL == 2) { accmax<2>(h2, mx); }
            else {
                quant_store<2>(h2, invs[2], act, g, nIdx);
                load_A(act, nIdx, g, Alo, Ahi);
                float h3[2][8];
                run_layer<2, true>(Alo, Ahi, B3, cs3, bq3, h3);
                if constexpr (NL == 3) { accmax<2>(h3, mx); }
                else {
                    quant_store<2>(h3, invs[3], act, g, nIdx);
                    load_A(act, nIdx, g, Alo, Ahi);
                    float h4[1][8];
                    run_layer<1, false>(Alo, Ahi, B4, cs4, bq4, h4);
                    // Stage logits in LDS, then 16 lanes do one 5-way softmax row each.
                    #pragma unroll
                    for (int i = 0; i < 8; ++i)
                        hbuf[(i + (g << 3)) * 16 + nIdx] = h4[0][i];
                    if (lane < 16) {
                        const float* hr = hbuf + lane * 16;
                        const float v0 = hr[0], v1 = hr[1], v2 = hr[2], v3 = hr[3], v4 = hr[4];
                        const float m = fmaxf(fmaxf(fmaxf(v0, v1), fmaxf(v2, v3)), v4);
                        const float e0 = __expf(v0 - m), e1 = __expf(v1 - m);
                        const float e2 = __expf(v2 - m), e3 = __expf(v3 - m);
                        const float e4 = __expf(v4 - m);
                        const float inv = 1.0f / (e0 + e1 + e2 + e3 + e4);
                        float* o = out + (size_t)(row0 + lane) * 5;
                        o[0] = e0 * inv; o[1] = e1 * inv; o[2] = e2 * inv;
                        o[3] = e3 * inv; o[4] = e4 * inv;
                    }
                }
            }
        }
    }

    if constexpr (NL < 4) block_max_atomic(mx, sred, &ws->aw[NL]);
}

// ---------------------------------------------------------------------------
extern "C" void kernel_launch(void* const* d_in, const int* in_sizes, int n_in,
                              void* d_out, int out_size, void* d_ws, size_t ws_size,
                              hipStream_t stream) {
    const float* x  = (const float*)d_in[0];
    const float* W1 = (const float*)d_in[1];
    const float* b1 = (const float*)d_in[2];
    const float* W2 = (const float*)d_in[3];
    const float* b2 = (const float*)d_in[4];
    const float* W3 = (const float*)d_in[5];
    const float* b3 = (const float*)d_in[6];
    const float* W4 = (const float*)d_in[7];
    const float* b4 = (const float*)d_in[8];
    float* out = (float*)d_out;
    WS* ws = (WS*)d_ws;                           // needs ~18.5 KB of scratch

    init_kernel<<<1, 32, 0, stream>>>(ws);
    prep_kernel<<<1, 256, 0, stream>>>(W1, b1, W2, b2, W3, b3, W4, b4, ws);
    absmax_kernel<<<1024, 256, 0, stream>>>((const float4*)x, (ROWS * 16) / 4, ws);
    fwd_kernel<1><<<2048, 256, 0, stream>>>(x, out, ws);   // -> aw[1]
    fwd_kernel<2><<<2048, 256, 0, stream>>>(x, out, ws);   // -> aw[2]
    fwd_kernel<3><<<2048, 256, 0, stream>>>(x, out, ws);   // -> aw[3]
    fwd_kernel<4><<<2048, 256, 0, stream>>>(x, out, ws);   // -> softmax out
}
